// PFNet7_44994077393269
// MI455X (gfx1250) — compile-verified, matrix operand order checked by reading, not run on hardware
//
#include <hip/hip_runtime.h>
#include <hip/hip_bf16.h>
#include <math.h>

typedef __attribute__((ext_vector_type(16))) _Float16 v16h;
typedef __attribute__((ext_vector_type(8)))  _Float16 v8h;
typedef __attribute__((ext_vector_type(8)))  float    v8f;
typedef int v4i __attribute__((vector_size(16)));

#define KNN   16
#define PROP  22

#if __has_builtin(__builtin_amdgcn_global_load_async_to_lds_b128)
#define HAVE_ASYNC_LDS 1
#else
#define HAVE_ASYNC_LDS 0
#endif

__device__ __forceinline__ void async_wait0() {
#if __has_builtin(__builtin_amdgcn_s_wait_asynccnt)
  __builtin_amdgcn_s_wait_asynccnt(0);
#else
  asm volatile("s_wait_asynccnt 0x0" ::: "memory");
#endif
}

// ---------------------------------------------------------------------------
// Generic fused GEMM, all operands pre-padded f16 so every load is an
// unconditional global_load_b128.
//   A  : [M][KPAD] f16 row-major (zero-padded past valid K)
//   Bt : [Npad][KPAD] f16 = transpose of W[K][N], zero-padded
//   out16: f16 [M][ld16] (pad cols written as 0)   or
//   out32: f32 [M][ldc]  (only n < Nv written)
// ACT: 0 none, 1 ELU, 2 leaky_relu(0.01)  (compile-time, branchless epilogue)
// Wave32: lane L -> row/col (L&15), K-halves kb=(L>>4)*8, per CDNA5 ISA.
// Inner loop: 4x global_load_b128 + 1x v_wmma_f32_16x16x32_f16, fully unrolled.
// ---------------------------------------------------------------------------
template <int KPAD, int ACT>
__global__ __launch_bounds__(256) void wgemm_kernel(
    const _Float16* __restrict__ A,
    const _Float16* __restrict__ Bt,
    const float* __restrict__ bias,
    _Float16* __restrict__ out16, int ld16,
    float* __restrict__ out32, int ldc,
    int M, int Nv) {
  const int lane = threadIdx.x & 31;
  const int wave = threadIdx.x >> 5;
  const int ntiles = (Nv + 15) >> 4;
  const int tile = blockIdx.x * 8 + wave;
  if (tile >= (M >> 4) * ntiles) return;
  const int m0 = (tile / ntiles) * 16;
  const int n0 = (tile % ntiles) * 16;

  const int kb = (lane >> 4) * 8;
  const _Float16* arow = A  + (size_t)(m0 + (lane & 15)) * KPAD + kb;
  const _Float16* brow = Bt + (size_t)(n0 + (lane & 15)) * KPAD + kb;

  v8f acc = {};
#pragma unroll
  for (int ks = 0; ks < KPAD / 32; ++ks) {
    const v8h a0 = *(const v8h*)(arow + ks * 32);
    const v8h a1 = *(const v8h*)(arow + ks * 32 + 16);
    const v8h b0 = *(const v8h*)(brow + ks * 32);
    const v8h b1 = *(const v8h*)(brow + ks * 32 + 16);
    const v16h a = __builtin_shufflevector(a0, a1, 0,1,2,3,4,5,6,7,8,9,10,11,12,13,14,15);
    const v16h b = __builtin_shufflevector(b0, b1, 0,1,2,3,4,5,6,7,8,9,10,11,12,13,14,15);
    acc = __builtin_amdgcn_wmma_f32_16x16x32_f16(
        false, a, false, b, (short)0, acc, false, false);
  }

  const int n = n0 + (lane & 15);
  const int mb = (lane >> 4) * 8;
  const bool nok = (n < Nv);
  const float bsv = nok ? bias[n] : 0.0f;
#pragma unroll
  for (int r = 0; r < 8; ++r) {
    float v = acc[r] + bsv;
    if (ACT == 1) v = (v > 0.0f) ? v : expm1f(v);
    if (ACT == 2) v = (v > 0.0f) ? v : 0.01f * v;
    const size_t m = (size_t)(m0 + mb + r);
    if (out16) out16[m * ld16 + n] = (_Float16)(nok ? v : 0.0f);
    if (out32 && nok) out32[m * ldc + n] = v;
  }
}

// ---------------------------------------------------------------------------
// Brute-force kNN in learned 4-D space (one thread = one query).
// Candidate chunks (1024 x float4 = 16 KB) staged into LDS via the CDNA5
// async DMA path (global_load_async_to_lds_b128, ASYNCcnt), norms derived
// in LDS; register-resident fully-predicated sorted top-16, ties -> lower
// index (matches lax.top_k).
// ---------------------------------------------------------------------------
__global__ __launch_bounds__(256) void knn_kernel(
    const float* __restrict__ s, int n,
    int* __restrict__ nbr,
    float* __restrict__ edge_index,
    float* __restrict__ edge_weight) {
  __shared__ __align__(16) float lsraw[1024 * 4];
  __shared__ float lnrm[1024];
  const int tid  = threadIdx.x;
  const int node = blockIdx.x * 256 + tid;

  const float4 q = *(const float4*)(s + (size_t)node * 4);
  const float q2 = q.x * q.x + q.y * q.y + q.z * q.z + q.w * q.w;

  float bd[KNN];
  int   bi[KNN];
#pragma unroll
  for (int j = 0; j < KNN; ++j) { bd[j] = INFINITY; bi[j] = 0x7fffffff; }

  for (int base = 0; base < n; base += 1024) {
    __syncthreads();
#if HAVE_ASYNC_LDS
#pragma unroll
    for (int t = 0; t < 4; ++t) {
      const int c = tid + t * 256;
      __builtin_amdgcn_global_load_async_to_lds_b128(
          (__attribute__((address_space(1))) v4i*)(void*)(s + (size_t)(base + c) * 4),
          (__attribute__((address_space(3))) v4i*)(void*)(lsraw + c * 4), 0, 0);
    }
    async_wait0();
    __syncthreads();
    for (int c = tid; c < 1024; c += 256) {
      const float4 v = *(const float4*)(lsraw + c * 4);
      lnrm[c] = v.x * v.x + v.y * v.y + v.z * v.z + v.w * v.w;
    }
#else
    for (int c = tid; c < 1024; c += 256) {
      const float4 v = *(const float4*)(s + (size_t)(base + c) * 4);
      *(float4*)(lsraw + c * 4) = v;
      lnrm[c] = v.x * v.x + v.y * v.y + v.z * v.z + v.w * v.w;
    }
#endif
    if (base + 1024 < n)  // next chunk toward L2 (global_prefetch_b8)
      __builtin_prefetch(s + (size_t)(base + 1024) * 4 + tid * 16, 0, 1);
    __syncthreads();

#pragma unroll 4
    for (int c = 0; c < 1024; ++c) {
      const float4 v = *(const float4*)(lsraw + c * 4);
      const float dot = q.x * v.x + q.y * v.y + q.z * v.z + q.w * v.w;
      const float d   = q2 + lnrm[c] - 2.0f * dot;
      const int cand  = base + c;
      if (d < bd[KNN - 1] || (d == bd[KNN - 1] && cand < bi[KNN - 1])) {
        bool cb[KNN];
#pragma unroll
        for (int j = 0; j < KNN; ++j)
          cb[j] = (d < bd[j]) || (d == bd[j] && cand < bi[j]);
#pragma unroll
        for (int j = KNN - 1; j > 0; --j)
          if (cb[j - 1]) { bd[j] = bd[j - 1]; bi[j] = bi[j - 1]; }
#pragma unroll
        for (int j = 0; j < KNN; ++j)
          if (cb[j] && (j == 0 || !cb[j - 1])) { bd[j] = d; bi[j] = cand; }
      }
    }
  }

#pragma unroll
  for (int k = 0; k < KNN; ++k) {
    const size_t e = (size_t)node * KNN + k;
    nbr[e]         = bi[k];
    edge_weight[e] = __expf(-10.0f * fmaxf(bd[k], 0.0f));
    edge_index[e]                   = (float)bi[k];
    edge_index[(size_t)n * KNN + e] = (float)node;
  }
}

// ---------------------------------------------------------------------------
// Gather h[idx]*w, weighted mean+max; write after_message (f32, d_out) and
// the f16 zero-padded [x12 | mean | max | 0] (64 wide) lin_out input.
// ---------------------------------------------------------------------------
__global__ __launch_bounds__(256) void agg_kernel(
    const int* __restrict__ nbr, const float* __restrict__ ew,
    const float* __restrict__ h, const _Float16* __restrict__ x12p,
    float* __restrict__ after_msg, _Float16* __restrict__ xcat, int n) {
  const int node = blockIdx.x * 256 + threadIdx.x;
  if (node >= n) return;
  float mean[PROP], mx[PROP];
#pragma unroll
  for (int p = 0; p < PROP; ++p) { mean[p] = 0.0f; mx[p] = -INFINITY; }
#pragma unroll
  for (int k = 0; k < KNN; ++k) {
    const size_t e = (size_t)node * KNN + k;
    const int j = nbr[e];
    const float w = ew[e];
    const float* hj = h + (size_t)j * PROP;
#pragma unroll
    for (int p = 0; p < PROP; ++p) {
      const float v = hj[p] * w;
      mean[p] += v;
      mx[p] = fmaxf(mx[p], v);
    }
  }
  float* am = after_msg + (size_t)node * (2 * PROP);
  _Float16* xc = xcat + (size_t)node * 64;
  const _Float16* xr = x12p + (size_t)node * 32;
#pragma unroll
  for (int i = 0; i < 12; ++i) xc[i] = xr[i];
#pragma unroll
  for (int p = 0; p < PROP; ++p) {
    const float m = mean[p] * (1.0f / KNN);
    am[p] = m;            xc[12 + p] = (_Float16)m;
    am[PROP + p] = mx[p]; xc[12 + PROP + p] = (_Float16)mx[p];
  }
#pragma unroll
  for (int i = 56; i < 64; ++i) xc[i] = (_Float16)0.0f;
}

// nn3 input (f16, padded to 96): [x0(12) | pred_ids(6) | x64(64) | 0(14)]
__global__ __launch_bounds__(256) void pack3_kernel(
    const float* __restrict__ x0, const float* __restrict__ pid,
    const _Float16* __restrict__ x64p, _Float16* __restrict__ xc, int n) {
  const int node = blockIdx.x * 256 + threadIdx.x;
  if (node >= n) return;
  _Float16* r = xc + (size_t)node * 96;
#pragma unroll
  for (int i = 0; i < 12; ++i) r[i] = (_Float16)x0[(size_t)node * 12 + i];
#pragma unroll
  for (int i = 0; i < 6; ++i) r[12 + i] = (_Float16)pid[(size_t)node * 6 + i];
#pragma unroll
  for (int i = 0; i < 64; ++i) r[18 + i] = x64p[(size_t)node * 64 + i];
#pragma unroll
  for (int i = 82; i < 96; ++i) r[i] = (_Float16)0.0f;
}

// x0 -> f16 [N][32], zero-padded
__global__ __launch_bounds__(256) void packx0_kernel(
    const float* __restrict__ x0, _Float16* __restrict__ xp, int n) {
  const int node = blockIdx.x * 256 + threadIdx.x;
  if (node >= n) return;
  const float* r = x0 + (size_t)node * 12;
  _Float16* o = xp + (size_t)node * 32;
#pragma unroll
  for (int i = 0; i < 12; ++i) o[i] = (_Float16)r[i];
#pragma unroll
  for (int i = 12; i < 32; ++i) o[i] = (_Float16)0.0f;
}

// W[Kv][Nv] f32 -> Wt[Npad][Kpad] f16 transposed, zero-padded
__global__ __launch_bounds__(256) void packw_kernel(
    const float* __restrict__ W, int Kv, int Nv,
    _Float16* __restrict__ Wt, int Kpad, int total) {
  const int i = blockIdx.x * 256 + threadIdx.x;
  if (i >= total) return;
  const int nn = i / Kpad, k = i - nn * Kpad;
  const float v = (k < Kv && nn < Nv) ? W[(size_t)k * Nv + nn] : 0.0f;
  Wt[i] = (_Float16)v;
}

__global__ __launch_bounds__(256) void i2f_kernel(const int* __restrict__ in,
                                                  float* __restrict__ out, int n) {
  const int i = blockIdx.x * 256 + threadIdx.x;
  if (i < n) out[i] = (float)in[i];
}

// ---------------------------------------------------------------------------
extern "C" void kernel_launch(void* const* d_in, const int* in_sizes, int n_in,
                              void* d_out, int out_size, void* d_ws, size_t ws_size,
                              hipStream_t stream) {
  (void)n_in; (void)out_size; (void)ws_size;
  const int N = in_sizes[0] / 12;   // 65536

  const float* x0       = (const float*)d_in[0];
  const int*   ygen_id  = (const int*)  d_in[1];
  const float* ygen     = (const float*)d_in[2];
  const int*   ycand_id = (const int*)  d_in[3];
  const float* ycand    = (const float*)d_in[4];
  const float* n1W0 = (const float*)d_in[5],  *n1b0 = (const float*)d_in[6];
  const float* n1W1 = (const float*)d_in[7],  *n1b1 = (const float*)d_in[8];
  const float* n1W2 = (const float*)d_in[9],  *n1b2 = (const float*)d_in[10];
  const float* Ws   = (const float*)d_in[11], *bs   = (const float*)d_in[12];
  const float* Wh   = (const float*)d_in[13], *bh   = (const float*)d_in[14];
  const float* Wo   = (const float*)d_in[15], *bo   = (const float*)d_in[16];
  const float* n2W0 = (const float*)d_in[17], *n2b0 = (const float*)d_in[18];
  const float* n2W1 = (const float*)d_in[19], *n2b1 = (const float*)d_in[20];
  const float* n2W2 = (const float*)d_in[21], *n2b2 = (const float*)d_in[22];
  const float* n2W3 = (const float*)d_in[23], *n2b3 = (const float*)d_in[24];
  const float* n3W0 = (const float*)d_in[25], *n3b0 = (const float*)d_in[26];
  const float* n3W1 = (const float*)d_in[27], *n3b1 = (const float*)d_in[28];
  const float* n3W2 = (const float*)d_in[29], *n3b2 = (const float*)d_in[30];
  const float* n3W3 = (const float*)d_in[31], *n3b3 = (const float*)d_in[32];

  // ---- output layout (tuple return order, flattened) ----
  float* out = (float*)d_out;
  const size_t O_PID  = 0;
  const size_t O_P4   = O_PID  + (size_t)N * 6;
  const size_t O_YGID = O_P4   + (size_t)N * 6;
  const size_t O_YG   = O_YGID + (size_t)N;
  const size_t O_YCID = O_YG   + (size_t)N * 6;
  const size_t O_YC   = O_YCID + (size_t)N;
  const size_t O_EI   = O_YC   + (size_t)N * 6;
  const size_t O_EW   = O_EI   + (size_t)2 * N * KNN;
  const size_t O_AM   = O_EW   + (size_t)N * KNN;
  const size_t O_BM   = O_AM   + (size_t)N * 2 * PROP;

  // ---- workspace layout: f32 / int region, then f16 arena ----
  float* sbuf = (float*)d_ws;                       // [N,4] f32
  int*   nbr  = (int*)(sbuf + (size_t)N * 4);       // [N,16] int
  _Float16* f16a = (_Float16*)(nbr + (size_t)N * KNN);
  _Float16* x0p   = f16a;                           // [N,32]
  _Float16* t0    = x0p   + (size_t)N * 32;         // [N,64]
  _Float16* t1    = t0    + (size_t)N * 64;         // [N,64]
  _Float16* x12p  = t1    + (size_t)N * 64;         // [N,32]
  _Float16* xcat  = x12p  + (size_t)N * 32;         // [N,64] (56 valid)
  _Float16* x64p  = xcat  + (size_t)N * 64;         // [N,64]
  _Float16* xcat3 = x64p  + (size_t)N * 64;         // [N,96] (82 valid)
  _Float16* b0    = xcat3 + (size_t)N * 96;         // [N,256]
  _Float16* b1    = b0    + (size_t)N * 256;        // [N,256]
  _Float16* wpk   = b1    + (size_t)N * 256;        // packed weights (<1MB)

  const int nb = (N + 255) / 256;

  // ---- pack all weights (transposed, padded f16) ----
  _Float16* wp = wpk;
  auto packw = [&](const float* W, int Kv, int Nv, int Kpad) -> _Float16* {
    const int Npad = ((Nv + 15) / 16) * 16;
    _Float16* dst = wp;
    const int total = Npad * Kpad;
    wp += total;
    packw_kernel<<<(total + 255) / 256, 256, 0, stream>>>(W, Kv, Nv, dst, Kpad, total);
    return dst;
  };
  _Float16* n1W0t = packw(n1W0, 12, 64, 32);
  _Float16* n1W1t = packw(n1W1, 64, 64, 64);
  _Float16* n1W2t = packw(n1W2, 64, 12, 64);
  _Float16* Wst   = packw(Ws, 12, 4, 32);
  _Float16* Wht   = packw(Wh, 12, PROP, 32);
  _Float16* Wot   = packw(Wo, 56, 64, 64);
  _Float16* n2W0t = packw(n2W0, 64, 256, 64);
  _Float16* n2W1t = packw(n2W1, 256, 256, 256);
  _Float16* n2W2t = packw(n2W2, 256, 256, 256);
  _Float16* n2W3t = packw(n2W3, 256, 6, 256);
  _Float16* n3W0t = packw(n3W0, 82, 256, 96);
  _Float16* n3W1t = packw(n3W1, 256, 256, 256);
  _Float16* n3W2t = packw(n3W2, 256, 256, 256);
  _Float16* n3W3t = packw(n3W3, 256, 6, 256);

  auto gemm = [&](int Kpad, int act, const _Float16* A, const _Float16* Bt,
                  const float* bias, _Float16* o16, int ld16, float* o32,
                  int ldc, int Nv) {
    const int tiles = (N / 16) * ((Nv + 15) / 16);
    const int blocks = (tiles + 7) / 8;
#define GEMM_CASE(KP, AC)                                                      \
    case (KP * 4 + AC):                                                        \
      wgemm_kernel<KP, AC><<<blocks, 256, 0, stream>>>(A, Bt, bias, o16, ld16, \
                                                       o32, ldc, N, Nv);       \
      break;
    switch (Kpad * 4 + act) {
      GEMM_CASE(32, 0)  GEMM_CASE(32, 1)  GEMM_CASE(32, 2)
      GEMM_CASE(64, 0)  GEMM_CASE(64, 1)  GEMM_CASE(64, 2)
      GEMM_CASE(96, 0)  GEMM_CASE(96, 1)  GEMM_CASE(96, 2)
      GEMM_CASE(256, 0) GEMM_CASE(256, 1) GEMM_CASE(256, 2)
    }
#undef GEMM_CASE
  };

  // passthrough outputs
  i2f_kernel<<<nb, 256, 0, stream>>>(ygen_id,  out + O_YGID, N);
  i2f_kernel<<<nb, 256, 0, stream>>>(ycand_id, out + O_YCID, N);
  (void)hipMemcpyAsync(out + O_YG, ygen,  (size_t)N * 6 * sizeof(float), hipMemcpyDeviceToDevice, stream);
  (void)hipMemcpyAsync(out + O_YC, ycand, (size_t)N * 6 * sizeof(float), hipMemcpyDeviceToDevice, stream);

  // x0 -> padded f16; zero x12p (its cols 16..31 are not covered by tiles)
  packx0_kernel<<<nb, 256, 0, stream>>>(x0, x0p, N);
  (void)hipMemsetAsync(x12p, 0, (size_t)N * 32 * sizeof(_Float16), stream);

  // nn1: 12 -> 64 -> 64 -> 12
  gemm(32, 1, x0p, n1W0t, n1b0, t0, 64, nullptr, 0, 64);
  gemm(64, 1, t0,  n1W1t, n1b1, t1, 64, nullptr, 0, 64);
  gemm(64, 0, t1,  n1W2t, n1b2, x12p, 32, nullptr, 0, 12);

  // GravNet projections: s (f32 for kNN), h (f32 "before_message" in d_out)
  gemm(32, 0, x12p, Wst, bs, nullptr, 0, sbuf, 4, 4);
  gemm(32, 0, x12p, Wht, bh, nullptr, 0, out + O_BM, PROP, PROP);

  // kNN in s-space -> nbr, edge_index, edge_weight
  knn_kernel<<<N / 256, 256, 0, stream>>>(sbuf, N, nbr, out + O_EI, out + O_EW);

  // weighted mean+max aggregation -> after_message + f16 [x,agg] concat
  agg_kernel<<<nb, 256, 0, stream>>>(nbr, out + O_EW, out + O_BM, x12p,
                                     out + O_AM, xcat, N);

  // lin_out: 56 -> 64, leaky_relu
  gemm(64, 2, xcat, Wot, bo, x64p, 64, nullptr, 0, 64);

  // nn2: 64 -> 256 -> 256 -> 256 -> 6 (pred_ids)
  gemm(64,  1, x64p, n2W0t, n2b0, b0, 256, nullptr, 0, 256);
  gemm(256, 1, b0,   n2W1t, n2b1, b1, 256, nullptr, 0, 256);
  gemm(256, 1, b1,   n2W2t, n2b2, b0, 256, nullptr, 0, 256);
  gemm(256, 0, b0,   n2W3t, n2b3, nullptr, 0, out + O_PID, 6, 6);

  // nn3: pack [x0|pred_ids|x64] then 82 -> 256 -> 256 -> 256 -> 6 (pred_p4)
  pack3_kernel<<<nb, 256, 0, stream>>>(x0, out + O_PID, x64p, xcat3, N);
  gemm(96,  1, xcat3, n3W0t, n3b0, b1, 256, nullptr, 0, 256);
  gemm(256, 1, b1,    n3W1t, n3b1, b0, 256, nullptr, 0, 256);
  gemm(256, 1, b0,    n3W2t, n3b2, b1, 256, nullptr, 0, 256);
  gemm(256, 0, b1,    n3W3t, n3b3, nullptr, 0, out + O_P4, 6, 6);
}
